// MixedArityTreeLSTM_64510408786337
// MI455X (gfx1250) — matrix operand order repaired and checked by reading, not compiled, not run on hardware
//
#include <hip/hip_runtime.h>
#include <hip/hip_bf16.h>

typedef __attribute__((ext_vector_type(16))) _Float16 v16h;
typedef __attribute__((ext_vector_type(8)))  _Float16 v8h;
typedef __attribute__((ext_vector_type(8)))  float    v8f;

#define HDIM 128
#define BATCH 256
#define NTREE 511   // 2^(D+1)-1, D=8
#define NINT  255   // internal nodes per tree

// ---- branch-free fast activations (v_exp_f32 + v_rcp_f32, no EXEC games) --
__device__ __forceinline__ float fast_rcp(float x) {
  return __builtin_amdgcn_rcpf(x);
}
__device__ __forceinline__ float sigf(float x) {
  return fast_rcp(1.0f + __expf(-x));
}
__device__ __forceinline__ float tanh_fast(float x) {
  float e = __expf(-2.0f * __builtin_fabsf(x));     // (0,1]
  float t = (1.0f - e) * fast_rcp(1.0f + e);
  return __builtin_copysignf(t, x);
}

// ---- WMMA fragment helpers (CDNA5 16x16x32 f16 layouts) -------------------
// A (16x32 f16): lanes 0-15 -> row M=lane, K = k0+0..15
//                lanes 16-31 -> row M=lane-16, K = k0+16..31
__device__ __forceinline__ v16h load_a_frag(const _Float16* __restrict__ row,
                                            int k0, int lane_hi) {
  const _Float16* p = row + k0 + lane_hi * 16;    // 16B aligned
  v8h lo = *(const v8h*)(p);
  v8h hi = *(const v8h*)(p + 8);
  v16h r;
#pragma unroll
  for (int i = 0; i < 8; ++i) { r[i] = lo[i]; r[i + 8] = hi[i]; }
  return r;
}

// B (32x16 f16): lanes 0-15 -> col N=lane,    K = kbase+0..15
//                lanes 16-31 -> col N=lane-16, K = kbase+16..31
// Weights pre-transposed to [gate][N][K] f16 -> 16 contiguous halves per lane.
__device__ __forceinline__ v16h load_b_frag(const _Float16* __restrict__ base,
                                            int col, int kbase, int ld) {
  return *(const v16h*)(base + (size_t)col * ld + kbase);  // 32B aligned
}

__device__ __forceinline__ v8f wmma_f16(v16h a, v16h b, v8f c) {
  return __builtin_amdgcn_wmma_f32_16x16x32_f16(false, a, false, b,
                                                (short)0, c, false, false);
}

__device__ __forceinline__ v8f splat8(float x) {
  v8f r;
#pragma unroll
  for (int i = 0; i < 8; ++i) r[i] = x;
  return r;
}

// ---- weight transpose+downconvert: dst[g][n][k] = (f16) src[g][k][n] ------
__global__ void w_transpose_f16_kernel(const float* __restrict__ src,
                                       _Float16* __restrict__ dst,
                                       int K, int N, int total) {
  int i = blockIdx.x * blockDim.x + threadIdx.x;
  if (i >= total) return;
  int g   = i / (K * N);
  int rem = i % (K * N);
  int n   = rem / K;
  int k   = rem % K;
  dst[i] = (_Float16)src[((size_t)g * K + k) * N + n];
}

// ---- leaf level: h = tanh(x @ W[3] + bW[3]), c = 0 ------------------------
__global__ __launch_bounds__(128) void tree_leaf_kernel(
    const int* __restrict__ tokens, const float* __restrict__ emb,
    const float* __restrict__ bW, const _Float16* __restrict__ WT,
    float* __restrict__ h_out, float* __restrict__ c_out) {
  constexpr int LDA = 136;                 // 128 + pad (bank-conflict free)
  __shared__ _Float16 Atile[16][LDA];
  __shared__ int s_tok[16];
  const int tid = threadIdx.x, mtile = blockIdx.x;

  if (tid < 16) {
    int r = mtile * 16 + tid;
    int b = r >> 8, n = r & 255;           // rows = b*256 + leaf
    s_tok[tid] = tokens[b * NTREE + 255 + n];
  }
  __syncthreads();
  for (int idx = tid; idx < 16 * 128; idx += 128) {
    int m = idx >> 7, k = idx & 127;
    Atile[m][k] = (_Float16)emb[(size_t)s_tok[m] * HDIM + k];
  }
  __syncthreads();

  const int lane = tid & 31, wave = tid >> 5;
  const int lane_hi = (lane & 16) ? 1 : 0;
  const _Float16* arow = &Atile[lane & 15][0];
  const _Float16* W3 = WT + (size_t)3 * HDIM * HDIM;

#pragma unroll
  for (int tt = 0; tt < 2; ++tt) {
    int t = wave * 2 + tt;
    int col = t * 16 + (lane & 15);
    v8f acc = splat8(bW[3 * HDIM + col]);  // bias folded into accumulator
#pragma unroll
    for (int s = 0; s < 4; ++s) {
      v16h a = load_a_frag(arow, s * 32, lane_hi);
      v16h bm = load_b_frag(W3, col, s * 32 + lane_hi * 16, HDIM);
      acc = wmma_f16(a, bm, acc);
    }
#pragma unroll
    for (int v = 0; v < 8; ++v) {
      int M = v + lane_hi * 8;
      size_t r = (size_t)(mtile * 16 + M) * HDIM + col;
      h_out[r] = tanh_fast(acc[v]);
      c_out[r] = 0.0f;
    }
  }
}

// ---- one internal tree level ----------------------------------------------
__global__ __launch_bounds__(128) void tree_level_kernel(
    const int* __restrict__ tokens, const int* __restrict__ arity,
    const float* __restrict__ emb,
    const float* __restrict__ bW, const float* __restrict__ bUbin,
    const float* __restrict__ bUun,
    const _Float16* __restrict__ WT, const _Float16* __restrict__ UbT,
    const _Float16* __restrict__ UuT,
    const float* __restrict__ h_prev, const float* __restrict__ c_prev,
    float* __restrict__ h_out, float* __restrict__ c_out,
    int cnt, int o2) {
  constexpr int KW = 384, LDA = 392;       // [x(128)|h_l(128)|h_r(128)] + pad
  __shared__ _Float16 Atile[16][LDA];
  __shared__ int s_tok[16], s_child[16], s_arity[16], s_outrow[16];
  const int tid = threadIdx.x, mtile = blockIdx.x;

  if (tid == 0) {                           // warm GL2 for streamed weights
    __builtin_prefetch(WT, 0, 3);
    __builtin_prefetch(UbT, 0, 3);
    __builtin_prefetch(UuT, 0, 3);
  }
  if (tid < 16) {
    int r = mtile * 16 + tid;
    int b = r / cnt, n = r % cnt;
    s_tok[tid]    = tokens[b * NTREE + o2 + n];
    s_child[tid]  = b * 2 * cnt + 2 * n;    // left child row in prev level
    s_arity[tid]  = arity[b * NINT + o2 + n];
    s_outrow[tid] = b * cnt + n;
  }
  __syncthreads();
  for (int idx = tid; idx < 16 * KW; idx += 128) {
    int m = idx / KW, k = idx % KW;
    float v;
    if (k < 128)      v = emb[(size_t)s_tok[m] * HDIM + k];
    else if (k < 256) v = h_prev[(size_t)s_child[m] * HDIM + (k - 128)];
    else              v = h_prev[(size_t)(s_child[m] + 1) * HDIM + (k - 256)];
    Atile[m][k] = (_Float16)v;
  }
  __syncthreads();

  const int lane = tid & 31, wave = tid >> 5;
  const int lane_hi = (lane & 16) ? 1 : 0;
  const _Float16* arow = &Atile[lane & 15][0];

#pragma unroll
  for (int tt = 0; tt < 2; ++tt) {
    int t = wave * 2 + tt;
    int col = t * 16 + (lane & 15);

    // biases folded into accumulator init (col is constant per lane)
    v8f accWx[4], accUb[5], accUu[4];
#pragma unroll
    for (int g = 0; g < 4; ++g) accWx[g] = splat8(bW[g * HDIM + col]);
#pragma unroll
    for (int g = 0; g < 5; ++g) accUb[g] = splat8(bUbin[g * HDIM + col]);
#pragma unroll
    for (int g = 0; g < 4; ++g) accUu[g] = splat8(bUun[g * HDIM + col]);

    // Wx[g] = x @ W[g] : A K-range [0,128)
#pragma unroll
    for (int s = 0; s < 4; ++s) {
      v16h a = load_a_frag(arow, s * 32, lane_hi);
#pragma unroll
      for (int g = 0; g < 4; ++g) {
        v16h bm = load_b_frag(WT + (size_t)g * HDIM * HDIM, col,
                              s * 32 + lane_hi * 16, HDIM);
        accWx[g] = wmma_f16(a, bm, accWx[g]);
      }
    }
    // Ub[g] = [h_l,h_r] @ Ubin[g] : A K-range [128,384); Uu shares [128,256)
#pragma unroll
    for (int s = 0; s < 8; ++s) {
      v16h a = load_a_frag(arow, 128 + s * 32, lane_hi);
#pragma unroll
      for (int g = 0; g < 5; ++g) {
        v16h bm = load_b_frag(UbT + (size_t)g * HDIM * 256, col,
                              s * 32 + lane_hi * 16, 256);
        accUb[g] = wmma_f16(a, bm, accUb[g]);
      }
      if (s < 4) {
#pragma unroll
        for (int g = 0; g < 4; ++g) {
          v16h bm = load_b_frag(UuT + (size_t)g * HDIM * HDIM, col,
                                s * 32 + lane_hi * 16, HDIM);
          accUu[g] = wmma_f16(a, bm, accUu[g]);
        }
      }
    }

    // elementwise gating in registers (C layout: N=lane&15, M=v+8*lane_hi)
#pragma unroll
    for (int v = 0; v < 8; ++v) {
      int M = v + lane_hi * 8;
      int child = s_child[M], ar = s_arity[M], orow = s_outrow[M];
      float cl = c_prev[(size_t)child * HDIM + col];
      float cr = c_prev[(size_t)(child + 1) * HDIM + col];
      float wx0 = accWx[0][v], wx1 = accWx[1][v];
      float wx2 = accWx[2][v], wx3 = accWx[3][v];
      // binary node
      float i_b = sigf(wx0 + accUb[0][v]);
      float f_l = sigf(wx1 + accUb[1][v]);
      float f_r = sigf(wx1 + accUb[2][v]);
      float o_b = sigf(wx2 + accUb[3][v]);
      float u_b = tanh_fast(wx3 + accUb[4][v]);
      float c_b = i_b * u_b + f_l * cl + f_r * cr;
      float h_b = o_b * tanh_fast(c_b);
      // unary node
      float i_u = sigf(wx0 + accUu[0][v]);
      float f_u = sigf(wx1 + accUu[1][v]);
      float o_u = sigf(wx2 + accUu[2][v]);
      float u_u = tanh_fast(wx3 + accUu[3][v]);
      float c_u = i_u * u_u + f_u * cl;
      float h_u = o_u * tanh_fast(c_u);
      float hh  = (ar == 1) ? h_b : h_u;
      float ccv = (ar == 1) ? c_b : c_u;
      h_out[(size_t)orow * HDIM + col] = hh;
      c_out[(size_t)orow * HDIM + col] = ccv;
    }
  }
}

extern "C" void kernel_launch(void* const* d_in, const int* in_sizes, int n_in,
                              void* d_out, int out_size, void* d_ws,
                              size_t ws_size, hipStream_t stream) {
  (void)in_sizes; (void)n_in; (void)out_size; (void)ws_size;
  const int*   tokens = (const int*)d_in[0];
  const int*   arity  = (const int*)d_in[1];
  const float* emb    = (const float*)d_in[2];
  const float* W      = (const float*)d_in[3];
  const float* bW     = (const float*)d_in[4];
  const float* Ubin   = (const float*)d_in[5];
  const float* bUbin  = (const float*)d_in[6];
  const float* Uun    = (const float*)d_in[7];
  const float* bUun   = (const float*)d_in[8];
  float* out = (float*)d_out;

  char* ws = (char*)d_ws;
  const size_t S = (size_t)BATCH * 256 * HDIM * sizeof(float);  // per h/c buf
  float* hA = (float*)(ws);
  float* cA = (float*)(ws + S);
  float* hB = (float*)(ws + 2 * S);
  float* cB = (float*)(ws + 3 * S);
  _Float16* WT  = (_Float16*)(ws + 4 * S);
  _Float16* UbT = (_Float16*)(ws + 4 * S + (size_t)4 * 128 * 128 * 2);
  _Float16* UuT = (_Float16*)(ws + 4 * S + (size_t)4 * 128 * 128 * 2
                                         + (size_t)5 * 256 * 128 * 2);

  // f32 -> f16 weight transposes ([g][K][N] -> [g][N][K])
  {
    int t1 = 4 * 128 * 128;
    w_transpose_f16_kernel<<<(t1 + 255) / 256, 256, 0, stream>>>(W, WT, 128, 128, t1);
    int t2 = 5 * 256 * 128;
    w_transpose_f16_kernel<<<(t2 + 255) / 256, 256, 0, stream>>>(Ubin, UbT, 256, 128, t2);
    int t3 = 4 * 128 * 128;
    w_transpose_f16_kernel<<<(t3 + 255) / 256, 256, 0, stream>>>(Uun, UuT, 128, 128, t3);
  }

  // leaves (depth 8): 256 leaves * 256 batch / 16 rows per block
  tree_leaf_kernel<<<(BATCH * 256) / 16, 128, 0, stream>>>(tokens, emb, bW, WT, hA, cA);

  float *hp = hA, *cp = cA, *hc = hB, *cc = cB;
  for (int l = 7; l >= 0; --l) {
    int cnt = 1 << l, o2 = cnt - 1;
    float* ho = (l == 0) ? out : hc;
    float* co = (l == 0) ? (out + BATCH * HDIM) : cc;
    int nblk = (BATCH * cnt) / 16;
    tree_level_kernel<<<nblk, 128, 0, stream>>>(tokens, arity, emb, bW, bUbin,
                                                bUun, WT, UbT, UuT, hp, cp, ho,
                                                co, cnt, o2);
    float* th = hp; hp = hc; hc = th;
    float* tc = cp; cp = cc; cc = tc;
  }
}
